// ZebPoseMAE_50440095924648
// MI455X (gfx1250) — matrix operand order repaired and checked by reading, not compiled, not run on hardware
//
#include <hip/hip_runtime.h>
#include <hip/hip_bf16.h>

typedef __attribute__((ext_vector_type(16))) _Float16 v16h;
typedef __attribute__((ext_vector_type(8)))  _Float16 v8h;
typedef __attribute__((ext_vector_type(8)))  float    v8f;

#define NM        256
#define T_FULL    120
#define V_FULL    13
#define T_KEEP    72
#define V_KEEP    7
#define N_JMASK   6
#define K_DIM     64
#define M_DIM     256
#define COLS      504      // T_KEEP * V_KEEP
#define COLS_PAD  512
#define TV        1560     // T_FULL * V_FULL
#define IDX_PER_N 1152     // 72 keepFrame + 72 destT + 504 keepJoint + 504 destV
#define LEAKY_S   0.1f

// leaky(x) = where(x>=0, x, 0.1x) == max(x, 0.1x)  (exact, incl. -0 / NaN)
__device__ __forceinline__ float leaky(float x) { return fmaxf(x, LEAKY_S * x); }

// ---------------------------------------------------------------------------
// Kernel 1: per-sample masking indices (stable argsorts) + inverse scatter maps
// ---------------------------------------------------------------------------
__global__ __launch_bounds__(128)
void k_indices(const float* __restrict__ fnoise,
               const float* __restrict__ jnoise,
               int* __restrict__ idx) {
  const int n   = blockIdx.x;
  const int tid = threadIdx.x;
  __shared__ float fn[T_FULL];
  __shared__ int   rnk[T_FULL];
  __shared__ int   fids[T_FULL];
  __shared__ int   invF[T_FULL];

  for (int t = tid; t < T_FULL; t += blockDim.x) fn[t] = fnoise[n * T_FULL + t];
  __syncthreads();
  // stable ascending rank
  for (int t = tid; t < T_FULL; t += blockDim.x) {
    float v = fn[t]; int r = 0;
    for (int j = 0; j < T_FULL; ++j)
      r += (fn[j] < v) || (fn[j] == v && j < t);
    rnk[t] = r;
  }
  __syncthreads();
  // frame_ids = [sorted kept frame ids, sorted masked frame ids]
  for (int t = tid; t < T_FULL; t += blockDim.x) {
    bool keep = rnk[t] < T_KEEP;
    int pos = 0;
    for (int j = 0; j < t; ++j) pos += ((rnk[j] < T_KEEP) == keep);
    fids[keep ? pos : (T_KEEP + pos)] = t;
  }
  __syncthreads();
  for (int t = tid; t < T_FULL; t += blockDim.x) invF[fids[t]] = t;
  __syncthreads();

  int* b = idx + n * IDX_PER_N;
  for (int s = tid; s < T_KEEP; s += blockDim.x) {
    b[s]          = fids[s];   // keepFrame: actual frame id of kept slot s
    b[T_KEEP + s] = invF[s];   // destT: output time t whose frame_ids[t] == s
  }
  // joint masking, one kept slot per thread
  for (int s = tid; s < T_KEEP; s += blockDim.x) {
    float jn[V_FULL]; int jr[V_FULL]; int comb[V_FULL];
    for (int v = 0; v < V_FULL; ++v)
      jn[v] = jnoise[(n * T_KEEP + s) * V_FULL + v];
    for (int v = 0; v < V_FULL; ++v) {
      int r = 0;
      for (int j = 0; j < V_FULL; ++j)
        r += (jn[j] < jn[v]) || (jn[j] == jn[v] && j < v);
      jr[v] = r;
    }
    // combined = stable argsort of mask flag: unmasked joints (rank>=6) first
    int c0 = 0;
    for (int v = 0; v < V_FULL; ++v) if (jr[v] >= N_JMASK) comb[c0++] = v;
    for (int v = 0; v < V_FULL; ++v) if (jr[v] <  N_JMASK) comb[c0++] = v;
    for (int vk = 0; vk < V_KEEP; ++vk)
      b[2 * T_KEEP + s * V_KEEP + vk] = comb[vk];              // keepJoint
    for (int v = 0; v < V_FULL; ++v) {
      int cv = comb[v];
      if (cv < V_KEEP) b[2 * T_KEEP + COLS + s * V_KEEP + cv] = v;  // destV
    }
  }
}

// ---------------------------------------------------------------------------
// Kernel 2: baseline decode term (w_dec · dec_pos_embed) + w_enc -> f16
// ---------------------------------------------------------------------------
__global__ __launch_bounds__(256)
void k_prep(const float* __restrict__ wdec,
            const float* __restrict__ dpe,
            const float* __restrict__ wenc,
            float* __restrict__ dpeProj,
            _Float16* __restrict__ wencH) {
  const int gid  = blockIdx.x * blockDim.x + threadIdx.x;
  const int nthr = gridDim.x * blockDim.x;
  if (gid < TV) {
    float a0 = 0.f, a1 = 0.f;
    for (int c = 0; c < M_DIM; ++c) {
      float d = dpe[c * TV + gid];
      a0 = fmaf(wdec[c], d, a0);
      a1 = fmaf(wdec[M_DIM + c], d, a1);
    }
    dpeProj[gid]      = a0;
    dpeProj[TV + gid] = a1;
  }
  for (int i = gid; i < M_DIM * K_DIM; i += nthr)
    wencH[i] = (_Float16)wenc[i];
}

// ---------------------------------------------------------------------------
// Kernel 3: one block per sample. Gather+in-proj -> H (LDS, f16),
// WMMA GEMM e = leaky(w_enc·H) fused with w_dec projection, scatter to out.
// ---------------------------------------------------------------------------
__global__ __launch_bounds__(256)
void k_main(const float* __restrict__ x,
            const float* __restrict__ pos,
            const float* __restrict__ win,
            const float* __restrict__ bin,
            const _Float16* __restrict__ wencH,
            const float* __restrict__ wdec,
            const int* __restrict__ idx,
            const float* __restrict__ dpeProj,
            float* __restrict__ out) {
  extern __shared__ _Float16 Hlds[];           // COLS_PAD * K_DIM  (64 KB)
  __shared__ float winS[K_DIM * 2];
  __shared__ float binS[K_DIM];
  __shared__ float wdS[2 * M_DIM];

  const int n   = blockIdx.x;
  const int tid = threadIdx.x;
  const int* b  = idx + n * IDX_PER_N;
  float* outN   = out + (size_t)n * 2 * TV;

  for (int i = tid; i < K_DIM * 2; i += 256) winS[i] = win[i];
  for (int i = tid; i < K_DIM;     i += 256) binS[i] = bin[i];
  for (int i = tid; i < 2 * M_DIM; i += 256) wdS[i]  = wdec[i];

  // Phase A: baseline output = w_dec · dec_pos_embed (masked slots keep this)
  for (int i = tid; i < 2 * TV; i += 256) outN[i] = dpeProj[i];
  __syncthreads();  // winS/binS ready

  // Phase B: build H[col][k] = leaky(w_in · (x + pos_embed) + b_in), f16
  for (int p = tid; p < COLS_PAD; p += 256) {
    if (p < COLS) {
      int s  = p / V_KEEP, vk = p % V_KEEP;
      int fr = b[s];
      int jo = b[2 * T_KEEP + s * V_KEEP + vk];
      int o0 = fr * V_FULL + jo;
      float x0 = x[(size_t)n * 2 * TV + o0]      + pos[o0];
      float x1 = x[(size_t)n * 2 * TV + TV + o0] + pos[TV + o0];
      #pragma unroll 8
      for (int k = 0; k < K_DIM; ++k) {
        float h = fmaf(winS[2 * k], x0, fmaf(winS[2 * k + 1], x1, binS[k]));
        Hlds[p * K_DIM + k] = (_Float16)leaky(h);
      }
    } else {
      for (int k = 0; k < K_DIM; ++k) Hlds[p * K_DIM + k] = (_Float16)0.f;
    }
  }
  __syncthreads();

  // Phase C/D: E = leaky(Wenc[256x64] x H[64x504]) via v_wmma_f32_16x16x32_f16,
  // fused with out[o][col] = sum_c wdec[o][c]*E[c][col], then scatter.
  const int  wv   = tid >> 5;
  const int  ln   = tid & 31;
  const int  lnlo = ln & 15;
  const bool hiH  = (ln >= 16);

  for (int nt = wv; nt < COLS_PAD / 16; nt += 8) {
    const int col  = nt * 16 + lnlo;
    const int boff = hiH ? 16 : 0;
    // B operand: K contiguous per lane (lanes 0-15: K=0..15, 16-31: K=16..31)
    v16h b0 = *(const v16h*)&Hlds[col * K_DIM + 0  + boff];  // K 0..31 chunk
    v16h b1 = *(const v16h*)&Hlds[col * K_DIM + 32 + boff];  // K 32..63 chunk

    float acc0 = 0.f, acc1 = 0.f;
    for (int mt = 0; mt < M_DIM / 16; ++mt) {
      const int arow = mt * 16 + lnlo;
      const int alo  = hiH ? 16 : 0;
      const int ahi  = hiH ? 24 : 8;
      const _Float16* ap = wencH + arow * K_DIM;
      v8h a00 = *(const v8h*)(ap + alo);
      v8h a01 = *(const v8h*)(ap + ahi);
      v8h a10 = *(const v8h*)(ap + 32 + alo);
      v8h a11 = *(const v8h*)(ap + 32 + ahi);
      v16h a0 = __builtin_shufflevector(a00, a01, 0,1,2,3,4,5,6,7,8,9,10,11,12,13,14,15);
      v16h a1 = __builtin_shufflevector(a10, a11, 0,1,2,3,4,5,6,7,8,9,10,11,12,13,14,15);

      v8f c = {};
      c = __builtin_amdgcn_wmma_f32_16x16x32_f16(false, a0, false, b0,
                                                 (short)0, c, false, false);
      c = __builtin_amdgcn_wmma_f32_16x16x32_f16(false, a1, false, b1,
                                                 (short)0, c, false, false);

      const int rbase = mt * 16 + (hiH ? 8 : 0);
      #pragma unroll
      for (int j = 0; j < 8; ++j) {
        float e = leaky(c[j]);
        acc0 = fmaf(wdS[rbase + j],          e, acc0);
        acc1 = fmaf(wdS[M_DIM + rbase + j],  e, acc1);
      }
    }
    // combine row halves (lane L and L^16 hold the same column)
    acc0 += __shfl_xor(acc0, 16, 32);
    acc1 += __shfl_xor(acc1, 16, 32);

    const int p = nt * 16 + lnlo;
    if (!hiH && p < COLS) {
      int s  = p / V_KEEP, vk = p % V_KEEP;
      int t  = b[T_KEEP + s];
      int v  = b[2 * T_KEEP + COLS + s * V_KEEP + vk];
      int tv = t * V_FULL + v;
      outN[tv]      = acc0 + dpeProj[tv];
      outN[TV + tv] = acc1 + dpeProj[TV + tv];
    }
  }
}

// ---------------------------------------------------------------------------
extern "C" void kernel_launch(void* const* d_in, const int* in_sizes, int n_in,
                              void* d_out, int out_size, void* d_ws, size_t ws_size,
                              hipStream_t stream) {
  (void)in_sizes; (void)n_in; (void)out_size; (void)ws_size;
  const float* x      = (const float*)d_in[0];  // (256,2,120,13,1)
  const float* fnoise = (const float*)d_in[1];  // (256,120)
  const float* jnoise = (const float*)d_in[2];  // (256,72,13)
  const float* pos    = (const float*)d_in[3];  // (1,2,120,13)
  const float* dpe    = (const float*)d_in[4];  // (1,256,120,13)
  const float* win    = (const float*)d_in[5];  // (64,2)
  const float* bin    = (const float*)d_in[6];  // (64,)
  const float* wenc   = (const float*)d_in[7];  // (256,64)
  const float* wdec   = (const float*)d_in[8];  // (2,256)
  float* out = (float*)d_out;                   // (256,2,120,13,1)

  // workspace layout
  int*      idxWS   = (int*)d_ws;                                   // NM*1152 ints
  float*    dpeProj = (float*)d_ws + (size_t)NM * IDX_PER_N;        // 2*TV floats
  _Float16* wencH   = (_Float16*)(dpeProj + 2 * TV);                // 256*64 halves

  k_indices<<<NM, 128, 0, stream>>>(fnoise, jnoise, idxWS);
  k_prep<<<8, 256, 0, stream>>>(wdec, dpe, wenc, dpeProj, wencH);
  k_main<<<NM, 256, COLS_PAD * K_DIM * sizeof(_Float16), stream>>>(
      x, pos, win, bin, wencH, wdec, idxWS, dpeProj, out);
}
